// RiskyObject_61074434949915
// MI455X (gfx1250) — matrix operand order repaired and checked by reading, not compiled, not run on hardware
//
#include <hip/hip_runtime.h>
#include <hip/hip_bf16.h>

// ---------------------------------------------------------------- constants
#define T_STEPS 200
#define NBOX    30
#define XDIM    4096
#define HDIM    512
#define G3      1536          // 3*HDIM
#define NWG     8             // persistent workgroups in the scan kernel
#define MROWS   6200          // 200*31 valid rows of x
#define MPAD    6208          // padded to a multiple of 16

typedef __attribute__((ext_vector_type(16))) __bf16 v16bf;
typedef __attribute__((ext_vector_type(8)))  __bf16 v8bf;
typedef __attribute__((ext_vector_type(8)))  float  v8f;
typedef __attribute__((ext_vector_type(4)))  unsigned int u32x4;
typedef __attribute__((ext_vector_type(8)))  unsigned int u32x8;

// ------------------------------------------------------- workspace layout (bytes)
constexpr size_t OFF_PHIT = 0;                                    // [512][4096] bf16 (phi_w^T)
constexpr size_t OFF_WIH0 = OFF_PHIT + (size_t)HDIM*XDIM*2;       // [1536][1024] bf16
constexpr size_t OFF_WHH0 = OFF_WIH0 + (size_t)G3*(2*HDIM)*2;     // [1536][512] bf16
constexpr size_t OFF_WIH1 = OFF_WHH0 + (size_t)G3*HDIM*2;         // [1536][512] bf16
constexpr size_t OFF_WHH1 = OFF_WIH1 + (size_t)G3*HDIM*2;         // [1536][512] bf16
constexpr size_t OFF_D1T  = OFF_WHH1 + (size_t)G3*HDIM*2;         // [64][512] bf16 (d1_w^T)
constexpr size_t OFF_ZROW = OFF_D1T  + (size_t)64*HDIM*2;         // [4096] f32 zeros
constexpr size_t OFF_CTR  = OFF_ZROW + (size_t)XDIM*4;            // 64 ints (barrier)
constexpr size_t OFF_H0G  = OFF_CTR  + 256;                       // [32][512] f32
constexpr size_t OFF_H1G  = OFF_H0G  + (size_t)32*HDIM*4;         // [32][512] f32
constexpr size_t OFF_XV   = OFF_H1G  + (size_t)32*HDIM*4;         // [6208][512] bf16
constexpr size_t OFF_GI0  = OFF_XV   + (size_t)MPAD*HDIM*2;       // [6000][1536] f32
constexpr size_t OFF_XBF  = OFF_GI0  + (size_t)6000*G3*4;         // [6208][4096] bf16 (x in bf16)

// prep index-space sizes
constexpr size_t N_PHIT = (size_t)HDIM*XDIM;
constexpr size_t N_WIH0 = (size_t)G3*(2*HDIM);
constexpr size_t N_WHH  = (size_t)G3*HDIM;
constexpr size_t N_D1T  = (size_t)64*HDIM;
constexpr size_t N_ZROW = XDIM;
constexpr size_t N_CTR  = 64;
constexpr size_t N_HG   = (size_t)32*HDIM;
constexpr size_t N_XBF  = (size_t)MPAD*XDIM;          // convert 6200 rows + zero 8 pad rows
constexpr size_t PREP_TOTAL = N_PHIT + N_WIH0 + 3*N_WHH + N_D1T + N_ZROW + N_CTR + N_HG + N_XBF;

// ---------------------------------------------------------------- helpers
__device__ __forceinline__ v8f wmma_bf16(v16bf a, v16bf b, v8f c) {
  // D = A(16x32 bf16) * B(32x16 bf16) + C(16x16 f32)
  return __builtin_amdgcn_wmma_f32_16x16x32_bf16(false, a, false, b, (short)0, c,
                                                 false, false);
}

// A-fragment (16x32) from a row-major bf16 source (ISA 7.12.2 16-bit A layout).
__device__ __forceinline__ v16bf load_a_bf16(const __bf16* __restrict__ row, int k0, int kh) {
  const __bf16* p0 = row + k0 + 8*kh;
  v8bf lo = *(const v8bf*)p0;
  v8bf hi = *(const v8bf*)(p0 + 16);
  v16bf a;
#pragma unroll
  for (int e = 0; e < 8; ++e) { a[e] = lo[e]; a[8+e] = hi[e]; }
  return a;
}

// A-fragment with on-the-fly f32 -> bf16 conversion (for the fp32 h-state).
__device__ __forceinline__ v16bf load_a_f32(const float* __restrict__ row, int k0, int kh) {
  const float* p0 = row + k0 + 8*kh;
  const float* p1 = p0 + 16;
  v16bf a;
#pragma unroll
  for (int e = 0; e < 8; ++e) { a[e] = (__bf16)p0[e]; a[8+e] = (__bf16)p1[e]; }
  return a;
}

// A-fragment from bf16 x_t with obj(k<512) / img(k>=512) split folded in.
__device__ __forceinline__ v16bf load_a_xt(const __bf16* __restrict__ pObj,
                                           const __bf16* __restrict__ pImg,
                                           int k0, int kh) {
  int s0 = k0 + 8*kh;
  int s1 = s0 + 16;
  const __bf16* p0 = (s0 < HDIM) ? (pObj + s0) : (pImg + (s0 - HDIM));
  const __bf16* p1 = (s1 < HDIM) ? (pObj + s1) : (pImg + (s1 - HDIM));
  v8bf lo = *(const v8bf*)p0;
  v8bf hi = *(const v8bf*)p1;
  v16bf a;
#pragma unroll
  for (int e = 0; e < 8; ++e) { a[e] = lo[e]; a[8+e] = hi[e]; }
  return a;
}

// B-fragment (32x16) from a weight stored as W[n][k] (row-major N x K, bf16):
// lane: col n = nbase + (lane&15); elems e<8 = W[n][kbase + 8*(lane>>4) + e], e>=8 = W[n][kbase + 16 + 8*(lane>>4) + e-8].
__device__ __forceinline__ v16bf load_b_w(const __bf16* __restrict__ w, int ldk,
                                          int nbase, int kbase, int lane) {
  const __bf16* p = w + (size_t)(nbase + (lane & 15))*ldk + kbase + ((lane >> 4) << 3);
  v8bf lo = *(const v8bf*)p;
  v8bf hi = *(const v8bf*)(p + 16);
  v16bf b;
#pragma unroll
  for (int e = 0; e < 8; ++e) { b[e] = lo[e]; b[8+e] = hi[e]; }
  return b;
}

__device__ __forceinline__ float sigm(float x) { return 1.0f / (1.0f + expf(-x)); }

// generic LDS pointer -> 32-bit LDS byte address (aperture low bits, §10.2)
__device__ __forceinline__ unsigned int lds_off(const void* p) {
  return (unsigned int)(unsigned long long)p;
}

// async global -> LDS DMA, 16 bytes per lane, tracked by ASYNCcnt (§15.18.3)
__device__ __forceinline__ void async_copy_b128(unsigned int lds_addr, const void* gaddr) {
  asm volatile("global_load_async_to_lds_b128 %0, %1, off"
               :: "v"(lds_addr), "v"(gaddr)
               : "memory");
}
__device__ __forceinline__ void wait_async0() {
  asm volatile("s_wait_asynccnt 0x0" ::: "memory");
}

// Tensor Data Mover: stage a 2D bf16 tile (rows x cols, row stride = cols)
// from global into LDS. 2-SGPR-group form (VADDR2/3 = NULL -> <=2D tensor).
__device__ __forceinline__ void tdm_load_2d_bf16(unsigned int lds_addr, const void* gptr,
                                                 unsigned int cols, unsigned int rows) {
  unsigned long long ga = (unsigned long long)gptr;
  u32x4 g0;
  g0[0] = 1u;                                               // count=1 (valid user D#)
  g0[1] = lds_addr;                                         // LDS dest (bytes)
  g0[2] = (unsigned int)ga;                                 // global_addr[31:0]
  g0[3] = ((unsigned int)(ga >> 32) & 0x01FFFFFFu)          // global_addr[56:32]
          | 0x80000000u;                                    // type=2 ("image")
  u32x8 g1;
  g1[0] = 0x00010000u;                                      // wg_mask=0, data_size=1 (2B)
  g1[1] = (cols & 0xFFFFu) << 16;                           // tensor_dim0[15:0]  (bits 63:48)
  g1[2] = ((cols >> 16) & 0xFFFFu) | ((rows & 0xFFFFu) << 16); // dim0 hi | tensor_dim1 lo
  g1[3] = ((rows >> 16) & 0xFFFFu) | ((cols & 0xFFFFu) << 16); // dim1 hi | tile_dim0
  g1[4] = rows & 0xFFFFu;                                   // tile_dim1 (tile_dim2=0)
  g1[5] = cols;                                             // tensor_dim0_stride[31:0]
  g1[6] = 0u;
  g1[7] = 0u;
  asm volatile("tensor_load_to_lds %0, %1" :: "s"(g0), "s"(g1) : "memory");
}

// device-scope phase barrier for the persistent scan kernel
__device__ __forceinline__ void gbarrier(int* ctr, int& epoch) {
  __syncthreads();
  __threadfence();
  ++epoch;
  if (threadIdx.x == 0) {
    __hip_atomic_fetch_add(ctr, 1, __ATOMIC_RELEASE, __HIP_MEMORY_SCOPE_AGENT);
    while (__hip_atomic_load(ctr, __ATOMIC_ACQUIRE, __HIP_MEMORY_SCOPE_AGENT) < NWG*epoch)
      __builtin_amdgcn_s_sleep(2);
  }
  __syncthreads();
}

// ---------------------------------------------------------------- prep kernel
__global__ void __launch_bounds__(256) prep_kernel(
    const float* __restrict__ x,
    const float* __restrict__ phi_w, const float* __restrict__ w_ih0,
    const float* __restrict__ w_hh0, const float* __restrict__ w_ih1,
    const float* __restrict__ w_hh1, const float* __restrict__ d1_w,
    unsigned char* __restrict__ ws) {
  __bf16* phiT = (__bf16*)(ws + OFF_PHIT);
  __bf16* wih0 = (__bf16*)(ws + OFF_WIH0);
  __bf16* whh0 = (__bf16*)(ws + OFF_WHH0);
  __bf16* wih1 = (__bf16*)(ws + OFF_WIH1);
  __bf16* whh1 = (__bf16*)(ws + OFF_WHH1);
  __bf16* d1T  = (__bf16*)(ws + OFF_D1T);
  float*  zrow = (float*)(ws + OFF_ZROW);
  int*    ctr  = (int*)  (ws + OFF_CTR);
  float*  h0g  = (float*)(ws + OFF_H0G);
  float*  h1g  = (float*)(ws + OFF_H1G);
  __bf16* xbf  = (__bf16*)(ws + OFF_XBF);

  size_t stride = (size_t)gridDim.x * blockDim.x;
  for (size_t idx = (size_t)blockIdx.x*blockDim.x + threadIdx.x; idx < PREP_TOTAL; idx += stride) {
    size_t r = idx;
    if (r < N_XBF) {            // x -> bf16, pad rows zeroed
      xbf[r] = (r < (size_t)MROWS*XDIM) ? (__bf16)x[r] : (__bf16)0.0f;
      continue;
    }
    r -= N_XBF;
    if (r < N_PHIT) {           // transpose: phiT[h][k] = phi_w[k][h]
      size_t h = r / XDIM, k = r % XDIM;
      phiT[r] = (__bf16)phi_w[k*HDIM + h];
      continue;
    }
    r -= N_PHIT;
    if (r < N_WIH0) { wih0[r] = (__bf16)w_ih0[r]; continue; }
    r -= N_WIH0;
    if (r < N_WHH)  { whh0[r] = (__bf16)w_hh0[r]; continue; }
    r -= N_WHH;
    if (r < N_WHH)  { wih1[r] = (__bf16)w_ih1[r]; continue; }
    r -= N_WHH;
    if (r < N_WHH)  { whh1[r] = (__bf16)w_hh1[r]; continue; }
    r -= N_WHH;
    if (r < N_D1T) {            // transpose: d1T[n][k] = d1_w[k][n]
      size_t n = r / HDIM, k = r % HDIM;
      d1T[r] = (__bf16)d1_w[k*64 + n];
      continue;
    }
    r -= N_D1T;
    if (r < N_ZROW) { zrow[r] = 0.0f; continue; }
    r -= N_ZROW;
    if (r < N_CTR)  { ctr[r] = 0; continue; }
    r -= N_CTR;
    h0g[r] = 0.0f; h1g[r] = 0.0f;
  }
}

// ---------------------------------------------------------------- GEMM 1: xv = relu(xbf @ phiT^T + b) -> bf16
// M = 6208, N = 512, K = 4096. Two-stage fragment pipeline (sets A/B alternate):
// each accumulator keeps one serial def-use chain -> in-place WMMA, no acc copies.
__global__ void __launch_bounds__(256) phi_gemm(
    const __bf16* __restrict__ xbf, const float* __restrict__ phi_b,
    const __bf16* __restrict__ phiT, __bf16* __restrict__ xv) {
  const int lane = threadIdx.x & 31, wave = threadIdx.x >> 5;
  const int msub = wave & 1, nsub = wave >> 1;        // 8 waves: 2 x 4
  const int mbase = blockIdx.x*32 + msub*16;          // grid.x = 194
  const int nbase = blockIdx.y*256 + nsub*64;         // grid.y = 2
  const int kh = lane >> 4;
  const __bf16* arow = xbf + (size_t)(mbase + (lane & 15))*XDIM;

  v8f a0 = {}, a1 = {}, a2 = {}, a3 = {};
  v16bf aA  = load_a_bf16(arow, 0, kh);
  v16bf b0A = load_b_w(phiT, XDIM, nbase +  0, 0, lane);
  v16bf b1A = load_b_w(phiT, XDIM, nbase + 16, 0, lane);
  v16bf b2A = load_b_w(phiT, XDIM, nbase + 32, 0, lane);
  v16bf b3A = load_b_w(phiT, XDIM, nbase + 48, 0, lane);
  v16bf aB  = load_a_bf16(arow, 32, kh);
  v16bf b0B = load_b_w(phiT, XDIM, nbase +  0, 32, lane);
  v16bf b1B = load_b_w(phiT, XDIM, nbase + 16, 32, lane);
  v16bf b2B = load_b_w(phiT, XDIM, nbase + 32, 32, lane);
  v16bf b3B = load_b_w(phiT, XDIM, nbase + 48, 32, lane);
#pragma unroll 1
  for (int k0 = 64; k0 < XDIM; k0 += 64) {
    if ((k0 & 255) == 0)
      __builtin_prefetch(phiT + (size_t)(nbase + (lane & 15))*XDIM + k0 + 512, 0, 1);
    // stage 1: consume set A (k0-64), refill with k0
    a0 = wmma_bf16(aA, b0A, a0);
    a1 = wmma_bf16(aA, b1A, a1);
    a2 = wmma_bf16(aA, b2A, a2);
    a3 = wmma_bf16(aA, b3A, a3);
    aA  = load_a_bf16(arow, k0, kh);
    b0A = load_b_w(phiT, XDIM, nbase +  0, k0, lane);
    b1A = load_b_w(phiT, XDIM, nbase + 16, k0, lane);
    b2A = load_b_w(phiT, XDIM, nbase + 32, k0, lane);
    b3A = load_b_w(phiT, XDIM, nbase + 48, k0, lane);
    // stage 2: consume set B (k0-32), refill with k0+32
    a0 = wmma_bf16(aB, b0B, a0);
    a1 = wmma_bf16(aB, b1B, a1);
    a2 = wmma_bf16(aB, b2B, a2);
    a3 = wmma_bf16(aB, b3B, a3);
    aB  = load_a_bf16(arow, k0 + 32, kh);
    b0B = load_b_w(phiT, XDIM, nbase +  0, k0 + 32, lane);
    b1B = load_b_w(phiT, XDIM, nbase + 16, k0 + 32, lane);
    b2B = load_b_w(phiT, XDIM, nbase + 32, k0 + 32, lane);
    b3B = load_b_w(phiT, XDIM, nbase + 48, k0 + 32, lane);
  }
  a0 = wmma_bf16(aA, b0A, a0);  a0 = wmma_bf16(aB, b0B, a0);
  a1 = wmma_bf16(aA, b1A, a1);  a1 = wmma_bf16(aB, b1B, a1);
  a2 = wmma_bf16(aA, b2A, a2);  a2 = wmma_bf16(aB, b2B, a2);
  a3 = wmma_bf16(aA, b3A, a3);  a3 = wmma_bf16(aB, b3B, a3);

  v8f acc[4] = {a0, a1, a2, a3};
#pragma unroll
  for (int s = 0; s < 4; ++s) {
    int col = nbase + 16*s + (lane & 15);
    float bias = phi_b[col];
#pragma unroll
    for (int v = 0; v < 8; ++v) {
      int mr = mbase + v + 8*kh;                 // ISA 16x16 f32 C/D layout
      float val = acc[s][v] + bias;
      val = val > 0.0f ? val : 0.0f;
      if (mr < MROWS) xv[(size_t)mr*HDIM + col] = (__bf16)val;
    }
  }
}

// ---------------------------------------------------------------- GEMM 2: gi0 = x_t @ w_ih0^T + b_ih0 -> f32
// M = 6000 (= 200*30), N = 1536, K = 1024, two-stage pipelined.
__global__ void __launch_bounds__(256) gi0_gemm(
    const __bf16* __restrict__ xv, const __bf16* __restrict__ wih0,
    const float* __restrict__ b_ih0, float* __restrict__ gi0) {
  const int lane = threadIdx.x & 31, wave = threadIdx.x >> 5;
  const int mt = blockIdx.x;                           // 0..374
  const int nb = blockIdx.y*256 + wave*32;             // grid.y = 6
  const int r  = mt*16 + (lane & 15);
  const int t  = r / NBOX, n = r % NBOX;
  const int kh = lane >> 4;
  const __bf16* pObj = xv + (size_t)(t*31 + n + 1)*HDIM;
  const __bf16* pImg = xv + (size_t)(t*31)*HDIM;

  v8f a0 = {}, a1 = {};
  v16bf aA  = load_a_xt(pObj, pImg, 0, kh);
  v16bf b0A = load_b_w(wih0, 2*HDIM, nb,      0, lane);
  v16bf b1A = load_b_w(wih0, 2*HDIM, nb + 16, 0, lane);
  v16bf aB  = load_a_xt(pObj, pImg, 32, kh);
  v16bf b0B = load_b_w(wih0, 2*HDIM, nb,      32, lane);
  v16bf b1B = load_b_w(wih0, 2*HDIM, nb + 16, 32, lane);
#pragma unroll 1
  for (int k0 = 64; k0 < 2*HDIM; k0 += 64) {
    a0 = wmma_bf16(aA, b0A, a0);
    a1 = wmma_bf16(aA, b1A, a1);
    aA  = load_a_xt(pObj, pImg, k0, kh);
    b0A = load_b_w(wih0, 2*HDIM, nb,      k0, lane);
    b1A = load_b_w(wih0, 2*HDIM, nb + 16, k0, lane);
    a0 = wmma_bf16(aB, b0B, a0);
    a1 = wmma_bf16(aB, b1B, a1);
    aB  = load_a_xt(pObj, pImg, k0 + 32, kh);
    b0B = load_b_w(wih0, 2*HDIM, nb,      k0 + 32, lane);
    b1B = load_b_w(wih0, 2*HDIM, nb + 16, k0 + 32, lane);
  }
  a0 = wmma_bf16(aA, b0A, a0);  a0 = wmma_bf16(aB, b0B, a0);
  a1 = wmma_bf16(aA, b1A, a1);  a1 = wmma_bf16(aB, b1B, a1);

#pragma unroll
  for (int s = 0; s < 2; ++s) {
    int col = nb + 16*s + (lane & 15);
    float bias = b_ih0[col];
    const v8f& c = s ? a1 : a0;
#pragma unroll
    for (int v = 0; v < 8; ++v) {
      int mr = mt*16 + v + 8*kh;                 // always < 6000
      gi0[(size_t)mr*G3 + col] = c[v] + bias;
    }
  }
}

// ---------------------------------------------------------------- scan kernel (persistent, NWG blocks)
// LDS: d1T tile (TDM-staged, 64KB) + hin0/hin1 (64KB each) + hid + loss + sel
constexpr size_t SCAN_SMEM = (size_t)64*HDIM*2
                           + (size_t)(32*HDIM*2 + 32*64 + 32)*sizeof(float)
                           + 32*sizeof(int);

__global__ void __launch_bounds__(256, 1) scan_kernel(
    const float* __restrict__ y, const float* __restrict__ gi0,
    const __bf16* __restrict__ whh0, const float* __restrict__ bhh0,
    const __bf16* __restrict__ wih1, const __bf16* __restrict__ whh1,
    const float* __restrict__ bih1, const float* __restrict__ bhh1,
    const __bf16* __restrict__ d1T, const float* __restrict__ d1b,
    const float* __restrict__ d2w, const float* __restrict__ d2b,
    const float* __restrict__ zrow,
    float* __restrict__ h0g, float* __restrict__ h1g,
    int* __restrict__ ctr, float* __restrict__ out) {
  extern __shared__ unsigned char smem[];
  __bf16* s_d1t = (__bf16*)smem;                // [64][512] bf16 (TDM-staged)
  float* s_hin0 = (float*)(smem + (size_t)64*HDIM*2);  // [32][512]
  float* s_hin1 = s_hin0 + 32*HDIM;             // [32][512]
  float* s_hid  = s_hin1 + 32*HDIM;             // [32][64]
  float* s_loss = s_hid  + 32*64;               // [32]
  int*   s_sel  = (int*)(s_loss + 32);          // [32]

  const int tid  = threadIdx.x;
  const int lane = tid & 31, wave = tid >> 5;
  const int wg   = blockIdx.x;
  const int gwave = wg*8 + wave;                // 0..63: one gate-tile task per wave
  const int mt = gwave >> 5, jt = gwave & 31;   // mt: row-tile (0..1), jt: h-col tile (0..31)
  const int mbase = mt*16, j = jt*16;
  const int kh = lane >> 4;
  const int col = j + (lane & 15);

  // ---- one-time TDM stage of the dense-head weight into LDS ------------
  if (wg == 0) {
    if (wave == 0) {
      tdm_load_2d_bf16(lds_off(s_d1t), d1T, HDIM, 64);
      __builtin_amdgcn_s_wait_tensorcnt(0);     // TENSORcnt drain (wave 0 only)
    }
    __syncthreads();                            // publish LDS tile to all waves
  }

  int epoch = 0;
  float loss_acc = 0.0f;

  for (int t = 0; t < T_STEPS; ++t) {
    // -- id matching (30x30 search) --------------------------------------
    if (tid < 32) {
      int sel = -1;
      if (tid < NBOX && t > 0) {
        float cid = y[(size_t)(t*NBOX + tid)*6];
        if (cid != 0.0f) {
          for (int p = 0; p < NBOX; ++p) {
            float pid = y[(size_t)((t-1)*NBOX + p)*6];
            if (pid == cid && pid != 0.0f) sel = p;
          }
        }
      }
      s_sel[tid] = sel;
    }
    __syncthreads();
    // -- gather h_in = match @ h_prev: async global->LDS DMA, 16B/lane ---
    for (int idx = tid; idx < 32*(HDIM/4); idx += 256) {
      int rr = idx >> 7;                 // row 0..31
      int c4 = (idx & 127) << 2;         // col in floats, 16B granular
      int p = s_sel[rr];
      const float* g0 = (p >= 0) ? (h0g + (size_t)p*HDIM + c4) : (zrow + c4);
      const float* g1 = (p >= 0) ? (h1g + (size_t)p*HDIM + c4) : (zrow + c4);
      async_copy_b128(lds_off(s_hin0 + (size_t)rr*HDIM + c4), g0);
      async_copy_b128(lds_off(s_hin1 + (size_t)rr*HDIM + c4), g1);
    }
    wait_async0();              // drain this wave's ASYNCcnt before the barrier
    gbarrier(ctr, epoch);       // all WGs gathered before anyone rewrites h*g

    // -- GRU layer 0: gh0 = hin0 @ w_hh0^T, fuse gates (2-stage pipeline) -
    {
      v8f ar = {}, az = {}, an = {};
      const float* arow = s_hin0 + (size_t)(mbase + (lane & 15))*HDIM;
      v16bf aA  = load_a_f32(arow, 0, kh);
      v16bf brA = load_b_w(whh0, HDIM, j,          0, lane);
      v16bf bzA = load_b_w(whh0, HDIM, HDIM + j,   0, lane);
      v16bf bnA = load_b_w(whh0, HDIM, 2*HDIM + j, 0, lane);
      v16bf aB  = load_a_f32(arow, 32, kh);
      v16bf brB = load_b_w(whh0, HDIM, j,          32, lane);
      v16bf bzB = load_b_w(whh0, HDIM, HDIM + j,   32, lane);
      v16bf bnB = load_b_w(whh0, HDIM, 2*HDIM + j, 32, lane);
#pragma unroll 1
      for (int k0 = 64; k0 < HDIM; k0 += 64) {
        ar = wmma_bf16(aA, brA, ar);
        az = wmma_bf16(aA, bzA, az);
        an = wmma_bf16(aA, bnA, an);
        aA  = load_a_f32(arow, k0, kh);
        brA = load_b_w(whh0, HDIM, j,          k0, lane);
        bzA = load_b_w(whh0, HDIM, HDIM + j,   k0, lane);
        bnA = load_b_w(whh0, HDIM, 2*HDIM + j, k0, lane);
        ar = wmma_bf16(aB, brB, ar);
        az = wmma_bf16(aB, bzB, az);
        an = wmma_bf16(aB, bnB, an);
        aB  = load_a_f32(arow, k0 + 32, kh);
        brB = load_b_w(whh0, HDIM, j,          k0 + 32, lane);
        bzB = load_b_w(whh0, HDIM, HDIM + j,   k0 + 32, lane);
        bnB = load_b_w(whh0, HDIM, 2*HDIM + j, k0 + 32, lane);
      }
      ar = wmma_bf16(aA, brA, ar);  ar = wmma_bf16(aB, brB, ar);
      az = wmma_bf16(aA, bzA, az);  az = wmma_bf16(aB, bzB, az);
      an = wmma_bf16(aA, bnA, an);  an = wmma_bf16(aB, bnB, an);

      float bhr = bhh0[col], bhz = bhh0[HDIM + col], bhn = bhh0[2*HDIM + col];
#pragma unroll
      for (int v = 0; v < 8; ++v) {
        int m = mbase + v + 8*kh;
        if (m < NBOX) {
          const float* gi = gi0 + (size_t)(t*NBOX + m)*G3;
          float r = sigm(gi[col] + ar[v] + bhr);
          float z = sigm(gi[HDIM + col] + az[v] + bhz);
          float n = tanhf(gi[2*HDIM + col] + r*(an[v] + bhn));
          float hb = s_hin0[(size_t)m*HDIM + col];
          h0g[(size_t)m*HDIM + col] = (1.0f - z)*n + z*hb;
        }
      }
    }
    gbarrier(ctr, epoch);       // h0 complete

    // -- GRU layer 1: gi1 = h0 @ w_ih1^T (+) gh1 = hin1 @ w_hh1^T --------
    {
      v8f ar = {}, az = {}, ani = {}, anh = {};
      const float* arow0 = h0g    + (size_t)(mbase + (lane & 15))*HDIM;
      const float* arow1 = s_hin1 + (size_t)(mbase + (lane & 15))*HDIM;
      {
        v16bf aA  = load_a_f32(arow0, 0, kh);
        v16bf brA = load_b_w(wih1, HDIM, j,          0, lane);
        v16bf bzA = load_b_w(wih1, HDIM, HDIM + j,   0, lane);
        v16bf bnA = load_b_w(wih1, HDIM, 2*HDIM + j, 0, lane);
        v16bf aB  = load_a_f32(arow0, 32, kh);
        v16bf brB = load_b_w(wih1, HDIM, j,          32, lane);
        v16bf bzB = load_b_w(wih1, HDIM, HDIM + j,   32, lane);
        v16bf bnB = load_b_w(wih1, HDIM, 2*HDIM + j, 32, lane);
#pragma unroll 1
        for (int k0 = 64; k0 < HDIM; k0 += 64) {
          ar  = wmma_bf16(aA, brA, ar);
          az  = wmma_bf16(aA, bzA, az);
          ani = wmma_bf16(aA, bnA, ani);
          aA  = load_a_f32(arow0, k0, kh);
          brA = load_b_w(wih1, HDIM, j,          k0, lane);
          bzA = load_b_w(wih1, HDIM, HDIM + j,   k0, lane);
          bnA = load_b_w(wih1, HDIM, 2*HDIM + j, k0, lane);
          ar  = wmma_bf16(aB, brB, ar);
          az  = wmma_bf16(aB, bzB, az);
          ani = wmma_bf16(aB, bnB, ani);
          aB  = load_a_f32(arow0, k0 + 32, kh);
          brB = load_b_w(wih1, HDIM, j,          k0 + 32, lane);
          bzB = load_b_w(wih1, HDIM, HDIM + j,   k0 + 32, lane);
          bnB = load_b_w(wih1, HDIM, 2*HDIM + j, k0 + 32, lane);
        }
        ar  = wmma_bf16(aA, brA, ar);  ar  = wmma_bf16(aB, brB, ar);
        az  = wmma_bf16(aA, bzA, az);  az  = wmma_bf16(aB, bzB, az);
        ani = wmma_bf16(aA, bnA, ani); ani = wmma_bf16(aB, bnB, ani);
      }
      {
        v16bf aA  = load_a_f32(arow1, 0, kh);
        v16bf brA = load_b_w(whh1, HDIM, j,          0, lane);
        v16bf bzA = load_b_w(whh1, HDIM, HDIM + j,   0, lane);
        v16bf bnA = load_b_w(whh1, HDIM, 2*HDIM + j, 0, lane);
        v16bf aB  = load_a_f32(arow1, 32, kh);
        v16bf brB = load_b_w(whh1, HDIM, j,          32, lane);
        v16bf bzB = load_b_w(whh1, HDIM, HDIM + j,   32, lane);
        v16bf bnB = load_b_w(whh1, HDIM, 2*HDIM + j, 32, lane);
#pragma unroll 1
        for (int k0 = 64; k0 < HDIM; k0 += 64) {
          ar  = wmma_bf16(aA, brA, ar);
          az  = wmma_bf16(aA, bzA, az);
          anh = wmma_bf16(aA, bnA, anh);
          aA  = load_a_f32(arow1, k0, kh);
          brA = load_b_w(whh1, HDIM, j,          k0, lane);
          bzA = load_b_w(whh1, HDIM, HDIM + j,   k0, lane);
          bnA = load_b_w(whh1, HDIM, 2*HDIM + j, k0, lane);
          ar  = wmma_bf16(aB, brB, ar);
          az  = wmma_bf16(aB, bzB, az);
          anh = wmma_bf16(aB, bnB, anh);
          aB  = load_a_f32(arow1, k0 + 32, kh);
          brB = load_b_w(whh1, HDIM, j,          k0 + 32, lane);
          bzB = load_b_w(whh1, HDIM, HDIM + j,   k0 + 32, lane);
          bnB = load_b_w(whh1, HDIM, 2*HDIM + j, k0 + 32, lane);
        }
        ar  = wmma_bf16(aA, brA, ar);  ar  = wmma_bf16(aB, brB, ar);
        az  = wmma_bf16(aA, bzA, az);  az  = wmma_bf16(aB, bzB, az);
        anh = wmma_bf16(aA, bnA, anh); anh = wmma_bf16(aB, bnB, anh);
      }
      float br  = bih1[col] + bhh1[col];
      float bz  = bih1[HDIM + col] + bhh1[HDIM + col];
      float bin = bih1[2*HDIM + col];
      float bhn = bhh1[2*HDIM + col];
#pragma unroll
      for (int v = 0; v < 8; ++v) {
        int m = mbase + v + 8*kh;
        if (m < NBOX) {
          float r = sigm(ar[v] + br);
          float z = sigm(az[v] + bz);
          float n = tanhf(ani[v] + bin + r*(anh[v] + bhn));
          float hb = s_hin1[(size_t)m*HDIM + col];
          h1g[(size_t)m*HDIM + col] = (1.0f - z)*n + z*hb;
        }
      }
    }
    gbarrier(ctr, epoch);       // h1 complete

    // -- dense head + loss (WG0 only; B-fragments from the TDM LDS tile) -
    if (wg == 0) {
      int dmt = wave & 1, dnt = wave >> 1;      // 8 waves: 2 x 4 tiles of hid(30x64)
      v8f acc = {};
      const float* arow = h1g + (size_t)(dmt*16 + (lane & 15))*HDIM;
      for (int k0 = 0; k0 < HDIM; k0 += 32) {
        v16bf a = load_a_f32(arow, k0, kh);
        acc = wmma_bf16(a, load_b_w(s_d1t, HDIM, dnt*16, k0, lane), acc);
      }
      int dcol = dnt*16 + (lane & 15);
      float bias = d1b[dcol];
#pragma unroll
      for (int v = 0; v < 8; ++v) {
        int m = dmt*16 + v + 8*kh;
        float hv = acc[v] + bias;
        s_hid[(size_t)m*64 + dcol] = hv > 0.0f ? hv : 0.0f;
      }
      __syncthreads();
      if (tid < NBOX) {
        float l0 = d2b[0], l1 = d2b[1];
        const float* hr = s_hid + (size_t)tid*64;
        for (int k = 0; k < 64; ++k) {
          float hv = hr[k];
          l0 += hv*d2w[k*2];
          l1 += hv*d2w[k*2 + 1];
        }
        float idv  = y[(size_t)(t*NBOX + tid)*6];
        float labf = y[(size_t)(t*NBOX + tid)*6 + 5];
        float af = (idv != 0.0f) ? 1.0f : 0.0f;
        float mx  = fmaxf(l0, l1);
        float lse = mx + logf(expf(l0 - mx) + expf(l1 - mx));
        float lp  = (((int)labf) == 0 ? l0 : l1) - lse;
        out[1 + (size_t)(t*NBOX + tid)*2 + 0] = l0*af;
        out[1 + (size_t)(t*NBOX + tid)*2 + 1] = l1*af;
        out[1 + (size_t)T_STEPS*NBOX*2 + (size_t)(t*NBOX + tid)] = labf;
        s_loss[tid] = -lp*af;
      }
      __syncthreads();
      if (tid == 0) {
        float s = 0.0f;
        for (int c = 0; c < NBOX; ++c) s += s_loss[c];
        loss_acc += s;
      }
    }
  }
  if (wg == 0 && tid == 0) out[0] = loss_acc;
}

// ---------------------------------------------------------------- launch
extern "C" void kernel_launch(void* const* d_in, const int* in_sizes, int n_in,
                              void* d_out, int out_size, void* d_ws, size_t ws_size,
                              hipStream_t stream) {
  (void)in_sizes; (void)n_in; (void)out_size; (void)ws_size;
  const float* x     = (const float*)d_in[0];
  const float* y     = (const float*)d_in[1];
  const float* phi_w = (const float*)d_in[2];
  const float* phi_b = (const float*)d_in[3];
  const float* w_ih0 = (const float*)d_in[4];
  const float* w_hh0 = (const float*)d_in[5];
  const float* b_ih0 = (const float*)d_in[6];
  const float* b_hh0 = (const float*)d_in[7];
  const float* w_ih1 = (const float*)d_in[8];
  const float* w_hh1 = (const float*)d_in[9];
  const float* b_ih1 = (const float*)d_in[10];
  const float* b_hh1 = (const float*)d_in[11];
  const float* d1_w  = (const float*)d_in[12];
  const float* d1_b  = (const float*)d_in[13];
  const float* d2_w  = (const float*)d_in[14];
  const float* d2_b  = (const float*)d_in[15];

  unsigned char* ws = (unsigned char*)d_ws;
  float* out = (float*)d_out;

  __bf16* phiT = (__bf16*)(ws + OFF_PHIT);
  __bf16* wih0 = (__bf16*)(ws + OFF_WIH0);
  __bf16* whh0 = (__bf16*)(ws + OFF_WHH0);
  __bf16* wih1 = (__bf16*)(ws + OFF_WIH1);
  __bf16* whh1 = (__bf16*)(ws + OFF_WHH1);
  __bf16* d1T  = (__bf16*)(ws + OFF_D1T);
  float*  zrow = (float*)(ws + OFF_ZROW);
  int*    ctr  = (int*)  (ws + OFF_CTR);
  float*  h0g  = (float*)(ws + OFF_H0G);
  float*  h1g  = (float*)(ws + OFF_H1G);
  __bf16* xv   = (__bf16*)(ws + OFF_XV);
  float*  gi0  = (float*)(ws + OFF_GI0);
  __bf16* xbf  = (__bf16*)(ws + OFF_XBF);

  prep_kernel<<<8192, 256, 0, stream>>>(x, phi_w, w_ih0, w_hh0, w_ih1, w_hh1, d1_w, ws);
  phi_gemm<<<dim3(194, 2), 256, 0, stream>>>(xbf, phi_b, phiT, xv);
  gi0_gemm<<<dim3(375, 6), 256, 0, stream>>>(xv, wih0, b_ih0, gi0);
  scan_kernel<<<NWG, 256, SCAN_SMEM, stream>>>(y, gi0, whh0, b_hh0, wih1, whh1,
                                               b_ih1, b_hh1, d1T, d1_b, d2_w, d2_b,
                                               zrow, h0g, h1g, ctr, out);
}
